// Graph_Convolution_5755256177264
// MI455X (gfx1250) — matrix operand order repaired and checked by reading, not compiled, not run on hardware
//
#include <hip/hip_runtime.h>
#include <hip/hip_bf16.h>

// GCN layer: h = X @ W ; out = adj @ h + b       (N=16384, F_IN=F_OUT=64)
//
// Roofline: adj is 1 GiB fp32, >> 192 MB L2, so it streams from HBM once:
// ~46 us at 23.3 TB/s. adj@h is 34.4 GFLOP -> needs ~750 TFLOP/s to stay
// bandwidth-bound; f32 WMMA (K=4/instr) is likely ~1/8 the bf16 rate, so we
// use bf16x3 split-precision (hi+lo RNE bf16, 3x v_wmma_f32_16x16x32_bf16,
// fp32 accum) -> ~16 mantissa bits, near-fp32 result, still HBM-bound.
// adj is loaded as fp32 (no extra traffic) and split in registers; h (4 MiB,
// L2-resident) is pre-split and stored transposed so B-fragments are
// contiguous 32 B per lane.

typedef __attribute__((ext_vector_type(16))) __bf16 v16bf;
typedef __attribute__((ext_vector_type(8)))  float  v8f;
typedef __attribute__((ext_vector_type(4)))  float  v4f;

typedef __attribute__((ext_vector_type(4))) unsigned int u32x4;
typedef __attribute__((ext_vector_type(8))) int          i32x8;
typedef __attribute__((ext_vector_type(4))) int          i32x4;

#define NN   16384
#define FIN  64
#define FOUT 64

#if __has_builtin(__builtin_amdgcn_tensor_load_to_lds) && \
    __has_builtin(__builtin_amdgcn_s_wait_tensorcnt)
#define HAVE_TDM 1
#else
#define HAVE_TDM 0
#endif

#if HAVE_TDM
// 1D fp32 tile copy global -> LDS via the Tensor Data Mover.
// D# built per CDNA5 ISA: group0 = {count=1, lds_addr,
// global_addr[56:0], type=2}; group1 = {wg_mask=0, data_size=4B(code 2),
// tensor_dim0=tile_elems (tensor declared == tile, so never OOB),
// tensor_dim1=1, tile_dim0=tile_elems, tile_dim1=1, stride=tile_elems}.
__device__ __forceinline__ void tdm_load_f32_1d(void* lds_dst,
                                                const float* gsrc,
                                                unsigned tile_elems) {
  const unsigned lds = (unsigned)(unsigned long long)lds_dst; // LDS byte offset
  const unsigned long long ga = (unsigned long long)gsrc;
  u32x4 g0;
  g0[0] = 1u;                                              // count=1, user D#
  g0[1] = lds;                                             // lds_addr
  g0[2] = (unsigned)(ga & 0xFFFFFFFFu);                    // global_addr lo
  g0[3] = (unsigned)((ga >> 32) & 0x1FFFFFFu) | (2u << 30);// addr hi | type=2
  i32x8 g1;
  g1[0] = 0x00020000;                                  // data_size=4B
  g1[1] = (int)((tile_elems & 0xFFFFu) << 16);         // tensor_dim0[15:0]
  g1[2] = (int)(((tile_elems >> 16) & 0xFFFFu)         // tensor_dim0[31:16]
                | (1u << 16));                         // tensor_dim1 = 1
  g1[3] = (int)((tile_elems & 0xFFFFu) << 16);         // tile_dim0
  g1[4] = 1;                                           // tile_dim1 = 1
  g1[5] = (int)tile_elems;                             // tensor_dim0_stride
  g1[6] = 0;
  g1[7] = 0;
  const i32x4 z4 = {0, 0, 0, 0};
#if __clang_major__ >= 23
  const i32x8 z8 = {0, 0, 0, 0, 0, 0, 0, 0};
  __builtin_amdgcn_tensor_load_to_lds(g0, g1, z4, z4, z8, 0);
#else
  __builtin_amdgcn_tensor_load_to_lds(g0, g1, z4, z4, 0);
#endif
}
#endif  // HAVE_TDM

// ---------------------------------------------------------------------------
// Kernel 1: h = input @ W (exact fp32), then split h into bf16 hi/lo pairs,
// stored TRANSPOSED (hT[f][node]) so WMMA B-fragments are contiguous per lane.
// W and the 64-node input tile are staged into LDS by the TDM (async tensor
// load + s_wait_tensorcnt), issued once by wave 0.
// ---------------------------------------------------------------------------
__global__ __launch_bounds__(256) void gcn_h_split_kernel(
    const float* __restrict__ input, const float* __restrict__ W,
    __bf16* __restrict__ hT_hi, __bf16* __restrict__ hT_lo)
{
  __shared__ float sW[FIN * FOUT];   // 16 KB
  __shared__ float sIn[64 * FIN];    // 16 KB (64 nodes per block)
  const int t = threadIdx.x;
  const int nodeBase = blockIdx.x * 64;

#if HAVE_TDM
  if (t < 32) {  // wave 0 issues both DMAs (TDM ignores EXEC; fires per wave)
    tdm_load_f32_1d(sW, W, FIN * FOUT);
    tdm_load_f32_1d(sIn, input + (size_t)nodeBase * FIN, 64 * FIN);
    __builtin_amdgcn_s_wait_tensorcnt(0);
  }
  __syncthreads();
#else
  const v4f* gW  = (const v4f*)W;
  const v4f* gIn = (const v4f*)(input + (size_t)nodeBase * FIN);
  for (int i = t; i < (FIN * FOUT) / 4; i += 256) {
    ((v4f*)sW)[i]  = gW[i];
    ((v4f*)sIn)[i] = gIn[i];
  }
  __syncthreads();
#endif

  const int col = t & 63;
  const int r0  = (t >> 6) * 16;     // 4 thread-groups x 16 rows = 64 rows
  for (int i = 0; i < 16; ++i) {
    const int row = r0 + i;
    float acc = 0.0f;
#pragma unroll
    for (int k = 0; k < FIN; ++k)
      acc = fmaf(sIn[row * FIN + k], sW[k * FOUT + col], acc);
    const __bf16 hi = (__bf16)acc;                   // RNE
    const __bf16 lo = (__bf16)(acc - (float)hi);     // residual
    const size_t idx = (size_t)col * NN + (size_t)(nodeBase + row);
    hT_hi[idx] = hi;
    hT_lo[idx] = lo;
  }
}

// ---------------------------------------------------------------------------
// Kernel 2: out = adj @ h + b via bf16x3 WMMA.
// Block = 128 threads = 4 waves; each wave owns one 16-row M-tile and all
// four 16-col N-tiles. K loop steps by 32. 256 blocks -> 1024 waves, each
// streaming 1 MiB of adj (16 rows).
//
// A-fragment (16x32 bf16): lane<16 -> M=lane, K=0..15;
// lane>=16 -> M=lane-16, K=16..31. Loaded as 4x b128 fp32
// (non-temporal: adj is stream-once), split to bf16 hi/lo in registers.
// B-fragment (32x16 bf16): lane<16 -> N=lane, K=0..15; lane>=16 -> K=16..31.
// With hT transposed this is 16 contiguous bf16 (32 B) per lane.
// ---------------------------------------------------------------------------
__global__ __launch_bounds__(128) void gcn_aggregate_kernel(
    const float* __restrict__ adj,
    const __bf16* __restrict__ hT_hi, const __bf16* __restrict__ hT_lo,
    const float* __restrict__ bias, float* __restrict__ out)
{
  const int lane  = threadIdx.x & 31;
  const int wave  = threadIdx.x >> 5;               // 0..3
  const int mtile = blockIdx.x * 4 + wave;          // 1024 M-tiles
  const int mbase = mtile * 16;

  const int halfSel = (lane < 16) ? 0 : 1;
  const int g0 = halfSel ? 16 : 0;                  // A K-group offset
  const int gb = halfSel ? 16 : 0;                  // B K-group offset
  const int ncol = lane & 15;

  const float* arowp = adj + (size_t)(mbase + ncol) * NN;  // row M = lane&15

  const __bf16* bph[4];
  const __bf16* bpl[4];
#pragma unroll
  for (int nt = 0; nt < 4; ++nt) {
    const size_t boff = (size_t)(nt * 16 + ncol) * NN + (size_t)gb;
    bph[nt] = hT_hi + boff;
    bpl[nt] = hT_lo + boff;
  }

  v8f acc[4] = {{}, {}, {}, {}};

  for (int k0 = 0; k0 < NN; k0 += 32) {
    // deep prefetch of this row's adj stream, 16 K-steps (2 KB) ahead;
    // wrap with & (NN-1) to avoid an OOB guard branch
    const int k0p = (k0 + 512) & (NN - 1);
    __builtin_prefetch(arowp + k0p + g0, 0, 0);
    __builtin_prefetch(arowp + k0p + g0 + 16, 0, 0);

    // ---- load A (fp32, stream-once -> non-temporal), split to hi/lo bf16
    v4f a[4];
    a[0] = __builtin_nontemporal_load((const v4f*)(arowp + k0 + g0));
    a[1] = __builtin_nontemporal_load((const v4f*)(arowp + k0 + g0 + 4));
    a[2] = __builtin_nontemporal_load((const v4f*)(arowp + k0 + g0 + 8));
    a[3] = __builtin_nontemporal_load((const v4f*)(arowp + k0 + g0 + 12));
    v16bf ahi, alo;
#pragma unroll
    for (int j = 0; j < 4; ++j) {
#pragma unroll
      for (int e = 0; e < 4; ++e) {
        const float f = a[j][e];
        const __bf16 hv = (__bf16)f;
        ahi[j * 4 + e] = hv;
        alo[j * 4 + e] = (__bf16)(f - (float)hv);
      }
    }
    // ---- load B fragments (L2/WGP$-resident, shared across waves) + 3x WMMA
#pragma unroll
    for (int nt = 0; nt < 4; ++nt) {
      const v16bf bh = *(const v16bf*)(bph[nt] + k0);
      const v16bf bl = *(const v16bf*)(bpl[nt] + k0);
      acc[nt] = __builtin_amdgcn_wmma_f32_16x16x32_bf16(
          false, ahi, false, bh, (short)0, acc[nt], false, false);
      acc[nt] = __builtin_amdgcn_wmma_f32_16x16x32_bf16(
          false, ahi, false, bl, (short)0, acc[nt], false, false);
      acc[nt] = __builtin_amdgcn_wmma_f32_16x16x32_bf16(
          false, alo, false, bh, (short)0, acc[nt], false, false);
    }
  }

  // ---- epilogue: C/D layout -> out[row][col] + bias
#pragma unroll
  for (int nt = 0; nt < 4; ++nt) {
    const int col = nt * 16 + ncol;
    const float bv = bias[col];
#pragma unroll
    for (int i = 0; i < 8; ++i) {
      const int row = mbase + i + (halfSel ? 8 : 0);
      out[(size_t)row * FOUT + col] = acc[nt][i] + bv;
    }
  }
}

// ---------------------------------------------------------------------------
extern "C" void kernel_launch(void* const* d_in, const int* in_sizes, int n_in,
                              void* d_out, int out_size, void* d_ws, size_t ws_size,
                              hipStream_t stream) {
  const float* input = (const float*)d_in[0];   // [16384, 64]
  const float* adj   = (const float*)d_in[1];   // [16384, 16384]
  const float* W     = (const float*)d_in[2];   // [64, 64]
  const float* bias  = (const float*)d_in[3];   // [64]
  float* out = (float*)d_out;                   // [16384, 64]

  // workspace: hT_hi / hT_lo, each [64][16384] bf16 (2 MiB each, 4 MiB total)
  __bf16* hT_hi = (__bf16*)d_ws;
  __bf16* hT_lo = hT_hi + (size_t)FOUT * NN;

  gcn_h_split_kernel<<<NN / 64, 256, 0, stream>>>(input, W, hT_hi, hT_lo);
  gcn_aggregate_kernel<<<NN / (16 * 4), 128, 0, stream>>>(adj, hT_hi, hT_lo,
                                                          bias, out);
}
